// PNAAggregation_67095979098788
// MI455X (gfx1250) — matrix operand order repaired and checked by reading, not compiled, not run on hardware
//
#include <hip/hip_runtime.h>
#include <hip/hip_bf16.h>
#include <math.h>

typedef __attribute__((ext_vector_type(16))) __bf16 v16bf;
typedef __attribute__((ext_vector_type(8)))  __bf16 v8bf;
typedef __attribute__((ext_vector_type(8)))  float  v8f;

#define IN_CH  32
#define OUT_CH 64
#define TOT    384   /* 4 aggregators x 3 scalers x 32 ch */
#define ROWPAD 392   /* LDS row stride (elements) to avoid bank conflicts */

// Order-preserving float<->int key map: signed-int compare == float compare.
__device__ __forceinline__ int f2key(float x) {
    int i = __float_as_int(x);
    return (i >= 0) ? i : (i ^ 0x7fffffff);
}
__device__ __forceinline__ float key2f(int k) {
    int i = (k >= 0) ? k : (k ^ 0x7fffffff);
    return __int_as_float(i);
}

__device__ __forceinline__ v16bf cat8(v8bf lo, v8bf hi) {
    return __builtin_shufflevector(lo, hi, 0,1,2,3,4,5,6,7,8,9,10,11,12,13,14,15);
}

__global__ void pna_init(float* __restrict__ sum, float* __restrict__ sumsq,
                         int* __restrict__ maxk, int* __restrict__ mink,
                         unsigned* __restrict__ cnt, int nn) {
    int i = blockIdx.x * blockDim.x + threadIdx.x;
    int total = nn * IN_CH;
    if (i < total) {
        sum[i]   = 0.0f;
        sumsq[i] = 0.0f;
        maxk[i]  = (int)0x80000000;  // below key(-inf)
        mink[i]  = 0x7fffffff;       // above key(+inf)
    }
    if (i < nn) cnt[i] = 0u;
}

__global__ void pna_wconv(const float* __restrict__ W, __bf16* __restrict__ Wbf, int n) {
    int i = blockIdx.x * blockDim.x + threadIdx.x;
    if (i < n) Wbf[i] = (__bf16)W[i];
}

// One thread per (edge, channel); wave == one edge => coalesced 128B message read.
__global__ void pna_scatter(const float* __restrict__ msgs, const int* __restrict__ dst,
                            float* __restrict__ sum, float* __restrict__ sumsq,
                            int* __restrict__ maxk, int* __restrict__ mink,
                            unsigned* __restrict__ cnt, int n_edges) {
    int idx = blockIdx.x * blockDim.x + threadIdx.x;
    int e = idx >> 5;
    int c = idx & 31;
    if (e >= n_edges) return;
    int node = dst[e];
    float v = msgs[(size_t)e * IN_CH + c];
    size_t o = (size_t)node * IN_CH + c;
    atomicAdd(&sum[o], v);
    atomicAdd(&sumsq[o], v * v);
    int k = f2key(v);
    atomicMax(&maxk[o], k);
    atomicMin(&mink[o], k);
    if (c == 0) atomicAdd(&cnt[node], 1u);
}

// Block = 128 threads (4 waves) = one 16-node tile. Fused feature build + WMMA GEMM.
__global__ __launch_bounds__(128) void pna_epilogue(
        const float* __restrict__ sum, const float* __restrict__ sumsq,
        const int* __restrict__ maxk, const int* __restrict__ mink,
        const unsigned* __restrict__ cnt, const __bf16* __restrict__ Wbf,
        const float* __restrict__ bias, float* __restrict__ out, int nn) {
    __shared__ __bf16 comb[16 * ROWPAD];

    const int tid  = threadIdx.x;
    const int base = blockIdx.x * 16;

    // Phase 1: build combined features [16 x 384] in LDS (bf16).
    for (int idx = tid; idx < 16 * IN_CH; idx += 128) {
        int r = idx >> 5, c = idx & 31;
        int node = base + r;
        float mean = 0.f, mx = 0.f, mn = 0.f, sd = 0.f, amp = 0.f, att = 0.f;
        if (node < nn) {
            float cn  = (float)cnt[node];
            float deg = fmaxf(cn, 1.0f);
            size_t o = (size_t)node * IN_CH + c;
            float s  = sum[o];
            float sq = sumsq[o];
            mean = s / deg;
            float var = fmaxf(sq / deg - mean * mean, 0.0f); // == E[(x-mean)^2]
            sd = sqrtf(var + 1e-8f);
            if (cn > 0.0f) { mx = key2f(maxk[o]); mn = key2f(mink[o]); }
            amp = logf(deg + 1.0f);
            att = 1.0f / (deg + 1.0f);
        }
        __bf16* row = &comb[r * ROWPAD];
        row[0*IN_CH + c]  = (__bf16)mean;
        row[1*IN_CH + c]  = (__bf16)(mean * amp);
        row[2*IN_CH + c]  = (__bf16)(mean * att);
        row[3*IN_CH + c]  = (__bf16)mx;
        row[4*IN_CH + c]  = (__bf16)(mx * amp);
        row[5*IN_CH + c]  = (__bf16)(mx * att);
        row[6*IN_CH + c]  = (__bf16)mn;
        row[7*IN_CH + c]  = (__bf16)(mn * amp);
        row[8*IN_CH + c]  = (__bf16)(mn * att);
        row[9*IN_CH + c]  = (__bf16)sd;
        row[10*IN_CH + c] = (__bf16)(sd * amp);
        row[11*IN_CH + c] = (__bf16)(sd * att);
    }
    __syncthreads();

    // Phase 2: each wave computes a 16(M) x 16(N) tile, K = 384 via 12 WMMAs.
    const int  lane    = tid & 31;
    const int  wave    = tid >> 5;
    const int  wstart  = wave * 16;          // output column base
    const int  n       = lane & 15;          // A: row M, B: col N
    const bool hiHalf  = lane >= 16;
    const int  kA      = hiHalf ? 8  : 0;    // A lanes 16-31 hold K 8..15 / 24..31
    const int  kB      = hiHalf ? 8  : 0;    // B lanes 16-31 hold K 8..15 / 24..31

    float bv = bias[wstart + n];             // bias depends only on column N
    v8f acc = {bv, bv, bv, bv, bv, bv, bv, bv};

    const __bf16* arow = &comb[n * ROWPAD];
    const __bf16* brow = Wbf + (size_t)(wstart + n) * TOT;  // B[K][N] = W[wstart+N][K]

    #pragma unroll
    for (int kc = 0; kc < TOT; kc += 32) {
        v8bf alo = *(const v8bf*)(arow + kc + kA);
        v8bf ahi = *(const v8bf*)(arow + kc + 16 + kA);
        v8bf blo = *(const v8bf*)(brow + kc + kB);
        v8bf bhi = *(const v8bf*)(brow + kc + kB + 16);
        v16bf a = cat8(alo, ahi);
        v16bf bm = cat8(blo, bhi);
        acc = __builtin_amdgcn_wmma_f32_16x16x32_bf16(
            /*neg_a=*/false, a, /*neg_b=*/false, bm,
            /*c_mod=*/(short)0, acc, /*reuse_a=*/false, /*reuse_b=*/false);
    }

    // D layout: VGPR v -> row v (lanes 0-15) / row v+8 (lanes 16-31), col = lane&15.
    #pragma unroll
    for (int v = 0; v < 8; ++v) {
        int mrow = hiHalf ? (v + 8) : v;
        int node = base + mrow;
        if (node < nn) out[(size_t)node * OUT_CH + wstart + n] = acc[v];
    }
}

extern "C" void kernel_launch(void* const* d_in, const int* in_sizes, int n_in,
                              void* d_out, int out_size, void* d_ws, size_t ws_size,
                              hipStream_t stream) {
    const float* msgs = (const float*)d_in[0];
    const int*   dst  = (const int*)d_in[1];
    // d_in[2] = num_nodes (device scalar; node count derived from out_size instead)
    const float* W    = (const float*)d_in[3];
    const float* bias = (const float*)d_in[4];
    float* out = (float*)d_out;

    const int n_edges = in_sizes[0] / IN_CH;
    const int nn      = out_size / OUT_CH;
    const size_t per  = (size_t)nn * IN_CH;

    // Workspace layout (~52 MB; fits entirely in 192 MB L2 during scatter)
    char* ws = (char*)d_ws;
    float*    sum   = (float*)ws;
    float*    sumsq = sum + per;
    int*      maxk  = (int*)(sumsq + per);
    int*      mink  = maxk + per;
    unsigned* cnt   = (unsigned*)(mink + per);
    size_t wOff = ((size_t)(4 * per + (size_t)nn) * 4 + 63) & ~(size_t)63;
    __bf16*   Wbf   = (__bf16*)(ws + wOff);
    (void)ws_size; (void)n_in;

    const int initTot = nn * IN_CH;
    pna_init<<<(initTot + 255) / 256, 256, 0, stream>>>(sum, sumsq, maxk, mink, cnt, nn);

    const int wTot = OUT_CH * TOT;
    pna_wconv<<<(wTot + 255) / 256, 256, 0, stream>>>(W, Wbf, wTot);

    const long sTot = (long)n_edges * IN_CH;
    pna_scatter<<<(int)((sTot + 255) / 256), 256, 0, stream>>>(
        msgs, dst, sum, sumsq, maxk, mink, cnt, n_edges);

    pna_epilogue<<<(nn + 15) / 16, 128, 0, stream>>>(
        sum, sumsq, maxk, mink, cnt, Wbf, bias, out, nn);
}
